// MMDMetric_82652350644520
// MI455X (gfx1250) — compile-verified, matrix operand order checked by reading, not run on hardware
//
#include <hip/hip_runtime.h>
#include <hip/hip_bf16.h>

typedef __attribute__((ext_vector_type(16))) _Float16 v16h;
typedef __attribute__((ext_vector_type(8)))  float    v8f;
typedef __attribute__((ext_vector_type(4)))  int      v4i;

#define NROWS 8192
#define DIMK  256
#define TILE  128
#define KSTEP 32

// ---- CDNA5 async global->LDS copy path (ASYNCcnt), with sync fallback ------
#if defined(__has_builtin)
#if __has_builtin(__builtin_amdgcn_global_load_async_to_lds_b128)
#define HAVE_ASYNC_LDS 1
#else
#define HAVE_ASYNC_LDS 0
#endif
#else
#define HAVE_ASYNC_LDS 0
#endif

// Builtin signature (from compiler diagnostic): param0 = v4i in addrspace(1)
// ("__device__"), param1 = v4i in addrspace(3), then imm offset + imm cpol.
#define GLOBAL_V4I(p) ((__attribute__((address_space(1))) v4i*)(p))
#define LDS_V4I(p)    ((__attribute__((address_space(3))) v4i*)(p))

__device__ __forceinline__ void wait_async_copies() {
#if HAVE_ASYNC_LDS
#if __has_builtin(__builtin_amdgcn_s_wait_asynccnt)
    __builtin_amdgcn_s_wait_asynccnt(0);
#else
    asm volatile("s_wait_asynccnt 0x0" ::: "memory");
#endif
#endif
}

// Stage one 128x32 f16 K-slab of P and Q into LDS (512 x 16B chunks per tile,
// 2 chunks per thread per tile).
__device__ __forceinline__ void stage_tiles(
    const _Float16* __restrict__ gP, const _Float16* __restrict__ gQ,
    _Float16* ldsA, _Float16* ldsB, int k0, int t)
{
    #pragma unroll
    for (int s = 0; s < 2; ++s) {
        const int c    = t + s * 256;             // 0..511
        const int row  = c >> 2;
        const int quad = c & 3;
        const size_t goff = (size_t)row * DIMK + k0 + quad * 8;
        const int    loff = row * KSTEP + quad * 8;
#if HAVE_ASYNC_LDS
        __builtin_amdgcn_global_load_async_to_lds_b128(
            GLOBAL_V4I(gP + goff), LDS_V4I(ldsA + loff), 0, 0);
        __builtin_amdgcn_global_load_async_to_lds_b128(
            GLOBAL_V4I(gQ + goff), LDS_V4I(ldsB + loff), 0, 0);
#else
        *(uint4*)(ldsA + loff) = *(const uint4*)(gP + goff);
        *(uint4*)(ldsB + loff) = *(const uint4*)(gQ + goff);
#endif
    }
}

// ---------------------------------------------------------------------------
// Prep: f32 -> f16 conversion, exact f32 row norms, zero the f64 accumulators.
// ---------------------------------------------------------------------------
__global__ __launch_bounds__(256) void mmd_prep(
    const float* __restrict__ real, const float* __restrict__ gen,
    _Float16* __restrict__ hR, _Float16* __restrict__ hG,
    float* __restrict__ nR, float* __restrict__ nG,
    double* __restrict__ accum)
{
    const int b = blockIdx.x;
    const int t = threadIdx.x;
    if (b == 0 && t < 3) accum[t] = 0.0;   // rr, rg, gg sums (main runs after)

    const int row = b & (NROWS - 1);
    const float*  src = (b < NROWS) ? real : gen;
    _Float16*     dst = (b < NROWS) ? hR   : hG;
    float*        nrm = (b < NROWS) ? nR   : nG;

    const float v = src[(size_t)row * DIMK + t];
    dst[(size_t)row * DIMK + t] = (_Float16)v;

    __shared__ float s[256];
    s[t] = v * v;
    __syncthreads();
    #pragma unroll
    for (int off = 128; off > 0; off >>= 1) {
        if (t < off) s[t] += s[t + off];
        __syncthreads();
    }
    if (t == 0) nrm[row] = s[0];
}

// ---------------------------------------------------------------------------
// Main: fused Gram + RBF + reduction. blockIdx.z = pair (0:rr, 1:rg, 2:gg).
// 128x128 tile per 256-thread block; 8 wave32s in a 2x4 grid, each wave owns
// a 64x32 sub-tile = 4x2 WMMA 16x16 fragments. Double-buffered LDS staging.
// ---------------------------------------------------------------------------
__global__ __launch_bounds__(256) void mmd_main(
    const _Float16* __restrict__ hR, const _Float16* __restrict__ hG,
    const float* __restrict__ nR, const float* __restrict__ nG,
    double* __restrict__ accum)
{
    const int pair = blockIdx.z;
    const _Float16* P  = (pair == 2) ? hG : hR;
    const _Float16* Q  = (pair == 0) ? hR : hG;
    const float*    nP = (pair == 2) ? nG : nR;
    const float*    nQ = (pair == 0) ? nR : nG;
    const bool sym = (pair != 1);

    const int ti = blockIdx.x;
    const int tj = blockIdx.y;
    if (sym && tj < ti) return;                       // upper-triangular tiles only
    const float weight = (sym && tj > ti) ? 2.0f : 1.0f;

    __shared__ _Float16 ldsA[2][TILE * KSTEP];        // 2 x 8 KB ping-pong
    __shared__ _Float16 ldsB[2][TILE * KSTEP];        // 2 x 8 KB ping-pong
    __shared__ float sNP[TILE];
    __shared__ float sNQ[TILE];
    __shared__ float sRed[256];

    const int t = threadIdx.x;
    if (t < TILE)       sNP[t]        = nP[ti * TILE + t];
    else                sNQ[t - TILE] = nQ[tj * TILE + (t - TILE)];

    const int lane = t & 31;
    const int wave = t >> 5;
    const int wrow = (wave & 1) * 64;                 // 0 / 64
    const int wcol = (wave >> 1) * 32;                // 0 / 32 / 64 / 96
    const int lrow = lane & 15;
    const int g    = lane >> 4;                       // half-wave group (K striping)

    v8f acc[4][2];
    #pragma unroll
    for (int r = 0; r < 4; ++r)
        #pragma unroll
        for (int c = 0; c < 2; ++c)
            acc[r][c] = (v8f){0.f,0.f,0.f,0.f,0.f,0.f,0.f,0.f};

    const _Float16* gP = P + (size_t)ti * TILE * DIMK;
    const _Float16* gQ = Q + (size_t)tj * TILE * DIMK;

    // Prologue: stage K-slab 0 into buffer 0.
    stage_tiles(gP, gQ, ldsA[0], ldsB[0], 0, t);
    wait_async_copies();
    __syncthreads();

    int buf = 0;
    for (int k0 = 0; k0 < DIMK; k0 += KSTEP, buf ^= 1) {
        // Overlap: kick off next slab's copy into the other buffer.
        if (k0 + KSTEP < DIMK)
            stage_tiles(gP, gQ, ldsA[buf ^ 1], ldsB[buf ^ 1], k0 + KSTEP, t);

        // Fragment loads: lane holds row M = lrow; halves 0-7 -> K=8g..8g+7,
        // halves 8-15 -> K=16+8g..16+8g+7 (ISA 16-bit A/B striping).
        v16h af[4], bf[2];
        #pragma unroll
        for (int r = 0; r < 4; ++r) {
            const _Float16* base = &ldsA[buf][(wrow + r * 16 + lrow) * KSTEP + g * 8];
            ((uint4*)&af[r])[0] = *(const uint4*)(base);
            ((uint4*)&af[r])[1] = *(const uint4*)(base + 16);
        }
        #pragma unroll
        for (int c = 0; c < 2; ++c) {
            const _Float16* base = &ldsB[buf][(wcol + c * 16 + lrow) * KSTEP + g * 8];
            ((uint4*)&bf[c])[0] = *(const uint4*)(base);
            ((uint4*)&bf[c])[1] = *(const uint4*)(base + 16);
        }
        #pragma unroll
        for (int r = 0; r < 4; ++r)
            #pragma unroll
            for (int c = 0; c < 2; ++c)
                acc[r][c] = __builtin_amdgcn_wmma_f32_16x16x32_f16(
                    false, af[r], false, bf[c], (short)0, acc[r][c], false, false);

        // Own async copies done, then block-wide handoff of the buffers.
        wait_async_copies();
        __syncthreads();
    }

    // Epilogue: C/D layout -> lane holds N = lrow, VGPR v holds M = v + 8*g.
    float lsum = 0.0f;
    #pragma unroll
    for (int r = 0; r < 4; ++r) {
        #pragma unroll
        for (int c = 0; c < 2; ++c) {
            #pragma unroll
            for (int v = 0; v < 8; ++v) {
                const int mi = wrow + r * 16 + v + 8 * g;
                const int nj = wcol + c * 16 + lrow;
                const float dot = acc[r][c][v];
                float d2 = sNP[mi] + sNQ[nj] - 2.0f * dot;
                d2 = fmaxf(d2, 1e-30f);
                float e = __expf(-sqrtf(d2) * (1.0f / (2.0f * DIMK)));
                // exact diagonal for rr/gg: d2==0 -> exp(-sqrt(EPS)/512)==1.0f
                if (sym && (ti * TILE + mi) == (tj * TILE + nj)) e = 1.0f;
                lsum += e;
            }
        }
    }

    sRed[t] = lsum;
    __syncthreads();
    #pragma unroll
    for (int off = 128; off > 0; off >>= 1) {
        if (t < off) sRed[t] += sRed[t + off];
        __syncthreads();
    }
    if (t == 0)
        atomicAdd(&accum[pair], (double)sRed[0] * (double)weight);
}

// ---------------------------------------------------------------------------
// Finalize: mmd = (rr + gg - 2*rg) / 8192^2
// ---------------------------------------------------------------------------
__global__ void mmd_final(const double* __restrict__ accum, float* __restrict__ out)
{
    const double m2 = (double)NROWS * (double)NROWS;
    out[0] = (float)((accum[0] + accum[2] - 2.0 * accum[1]) / m2);
}

extern "C" void kernel_launch(void* const* d_in, const int* in_sizes, int n_in,
                              void* d_out, int out_size, void* d_ws, size_t ws_size,
                              hipStream_t stream) {
    const float* real = (const float*)d_in[0];
    const float* gen  = (const float*)d_in[1];
    float* out = (float*)d_out;

    char* ws = (char*)d_ws;
    const size_t halfBytes = (size_t)NROWS * DIMK * sizeof(_Float16);  // 4 MB
    _Float16* hR = (_Float16*)(ws);
    _Float16* hG = (_Float16*)(ws + halfBytes);
    float*    nR = (float*)(ws + 2 * halfBytes);
    float*    nG = nR + NROWS;
    double*   accum = (double*)(ws + 2 * halfBytes + 2 * NROWS * sizeof(float));

    mmd_prep<<<2 * NROWS, 256, 0, stream>>>(real, gen, hR, hG, nR, nG, accum);

    dim3 grid(NROWS / TILE, NROWS / TILE, 3);   // 64 x 64 x {rr, rg, gg}
    mmd_main<<<grid, 256, 0, stream>>>(hR, hG, nR, nG, accum);

    mmd_final<<<1, 1, 0, stream>>>(accum, out);
}
